// AttnHeads_89653147337002
// MI455X (gfx1250) — compile-verified, matrix operand order checked by reading, not run on hardware
//
#include <hip/hip_runtime.h>
#include <hip/hip_bf16.h>

typedef _Float16 v8h  __attribute__((ext_vector_type(8)));
typedef _Float16 v16h __attribute__((ext_vector_type(16)));
typedef float    v8f  __attribute__((ext_vector_type(8)));

// Build a 16x32-K f16 WMMA operand fragment for this lane from a row base
// pointer (row-major, K contiguous). Lane half h owns K = {8h..8h+7, 16+8h..23+8h}.
__device__ __forceinline__ v16h frag16x32(const _Float16* rowBase, int h) {
    v8h lo = *(const v8h*)(rowBase + h * 8);
    v8h hi = *(const v8h*)(rowBase + 16 + h * 8);
    return __builtin_shufflevector(lo, hi, 0,1,2,3,4,5,6,7,8,9,10,11,12,13,14,15);
}

__device__ __forceinline__ v8f wmma_f16(v16h a, v16h b, v8f c) {
    return __builtin_amdgcn_wmma_f32_16x16x32_f16(false, a, false, b, (short)0, c, false, false);
}

// ---------------------------------------------------------------------------
// Kernel 1: fused QKV projection.  X[8192,1024] (f32) @ W[1024,3072] (f32)
// -> q/k/v f16 buffers laid out [b][n][s][d].
// Block tile 64(M) x 256(N), 8 waves (2x4), wave tile 32x64 (2x4 WMMA tiles),
// K chunks of 32.  8 WMMAs per 6 fragment loads per wave per K-step.
// ---------------------------------------------------------------------------
#define LDA 40   // halves: 32 + 8 pad (80B rows, 16B aligned)
#define LDB 40

__global__ __launch_bounds__(256) void qkv_gemm_kernel(
    const float* __restrict__ X, const float* __restrict__ W,
    _Float16* __restrict__ qb, _Float16* __restrict__ kb, _Float16* __restrict__ vb)
{
    __shared__ _Float16 As[64 * LDA];    // 5120 B
    __shared__ _Float16 Bts[256 * LDB];  // 20480 B, B tile transposed [n][k]

    const int tid  = threadIdx.x;
    const int wave = tid >> 5, lane = tid & 31;
    const int l15  = lane & 15, h = lane >> 4;
    const int wm   = wave >> 2, wn = wave & 3;
    const int m0   = blockIdx.y * 64;
    const int n0   = blockIdx.x * 256;

    v8f acc[2][4] = {};

    for (int k0 = 0; k0 < 1024; k0 += 32) {
        // Prefetch next K-tile into GL2 while we still compute on this one.
        if (k0 + 32 < 1024) {
            if (tid < 64)
                __builtin_prefetch(&X[(size_t)(m0 + tid) * 1024 + k0 + 32], 0, 0);
            __builtin_prefetch(&W[(size_t)(k0 + 32 + (tid >> 3)) * 3072 + n0 + (tid & 7) * 32], 0, 0);
        }
        __syncthreads();
#pragma unroll
        for (int e = 0; e < 8; ++e) {                 // A: 64x32 f32 -> f16
            int lin = tid + e * 256;
            int r = lin >> 5, c = lin & 31;
            As[r * LDA + c] = (_Float16)X[(size_t)(m0 + r) * 1024 + k0 + c];
        }
#pragma unroll
        for (int e = 0; e < 32; ++e) {                // B: 32x256 -> transpose
            int lin = tid + e * 256;                  // = k*256 + n (coalesced)
            int kk = lin >> 8, n = lin & 255;
            Bts[n * LDB + kk] = (_Float16)W[(size_t)(k0 + kk) * 3072 + n0 + n];
        }
        __syncthreads();

        v16h a0 = frag16x32(&As[(wm * 32 + l15) * LDA], h);
        v16h a1 = frag16x32(&As[(wm * 32 + 16 + l15) * LDA], h);
#pragma unroll
        for (int tj = 0; tj < 4; ++tj) {
            v16h bf = frag16x32(&Bts[(wn * 64 + tj * 16 + l15) * LDB], h);
            acc[0][tj] = wmma_f16(a0, bf, acc[0][tj]);
            acc[1][tj] = wmma_f16(a1, bf, acc[1][tj]);
        }
    }

#pragma unroll
    for (int ti = 0; ti < 2; ++ti)
#pragma unroll
        for (int tj = 0; tj < 4; ++tj)
#pragma unroll
            for (int r = 0; r < 8; ++r) {
                int row = m0 + wm * 32 + ti * 16 + h * 8 + r;   // b*2048 + s
                int col = n0 + wn * 64 + tj * 16 + l15;          // 0..3071
                int which = col >> 10;
                int hh = col & 1023, nh = hh >> 6, d = hh & 63;
                int b = row >> 11, s = row & 2047;
                _Float16* dst = (which == 0) ? qb : (which == 1) ? kb : vb;
                dst[((size_t)(b * 16 + nh) * 2048 + s) * 64 + d] = (_Float16)acc[ti][tj][r];
            }
}

// ---------------------------------------------------------------------------
// Kernel 2: flash attention per head.  score[t,s] = q_t . k_s (unscaled),
// softmax over s, ctx[t,d] = sum_s p[t,s] v[s,d].  Output ctx f16 [b*t][n*d].
// Block = 8 waves; wave owns 16 query rows; key chunks of 64 ->
// 16 WMMAs per wave between each pair of barriers.
// ---------------------------------------------------------------------------
#define LDS2 72  // shared row stride in halves for 64-wide tiles (144B, 16B mult)

__global__ __launch_bounds__(256) void flash_attn_kernel(
    const _Float16* __restrict__ qb, const _Float16* __restrict__ kbuf,
    const _Float16* __restrict__ vbuf, _Float16* __restrict__ ctxb)
{
    const int blk  = blockIdx.x;
    const int tt   = blk & 15;          // query tile of 128
    const int head = blk >> 4;          // b*16 + n
    const int bi   = head >> 4, nh = head & 15;
    const int tid  = threadIdx.x;
    const int wave = tid >> 5, lane = tid & 31;
    const int l15  = lane & 15, h = lane >> 4;

    const _Float16* Q = qb   + (size_t)head * 2048 * 64;
    const _Float16* K = kbuf + (size_t)head * 2048 * 64;
    const _Float16* V = vbuf + (size_t)head * 2048 * 64;
    const int t0 = tt * 128 + wave * 16;

    __shared__ _Float16 Ks[64 * LDS2];        // 9216 B, row-major [s][d]
    __shared__ _Float16 Vt[64 * LDS2];        // 9216 B, transposed [d][s]
    __shared__ _Float16 Ps[8 * 16 * LDS2];    // 18432 B, per-wave P tiles (16x64)
    _Float16* Pw = &Ps[wave * 16 * LDS2];

    // Resident Q fragments (rows t0..t0+15, two d-chunks) direct from global.
    v16h qa[2];
#pragma unroll
    for (int c = 0; c < 2; ++c)
        qa[c] = frag16x32(Q + (size_t)(t0 + l15) * 64 + c * 32, h);

    v8f cacc[4] = {};
    float rmax[8], rsum[8];
#pragma unroll
    for (int r = 0; r < 8; ++r) { rmax[r] = -__builtin_inff(); rsum[r] = 0.f; }

    for (int sc = 0; sc < 2048; sc += 64) {
        // Prefetch next key/value chunk (one 128B line per row per group of four threads).
        if (sc + 64 < 2048) {
            int pr = sc + 64 + (tid >> 2);
            __builtin_prefetch(K + (size_t)pr * 64, 0, 0);
            __builtin_prefetch(V + (size_t)pr * 64, 0, 0);
        }
        __syncthreads();
#pragma unroll
        for (int e = 0; e < 16; ++e) {       // stage K row-major, V transposed
            int lin = tid + e * 256;         // = s*64 + d (coalesced)
            int s = lin >> 6, d = lin & 63;
            _Float16 kv = K[(size_t)(sc + s) * 64 + d];
            _Float16 vv = V[(size_t)(sc + s) * 64 + d];
            Ks[s * LDS2 + d] = kv;
            Vt[d * LDS2 + s] = vv;
        }
        __syncthreads();

        // Scores: 16 queries x 64 keys = four 16x16 tiles, contraction d=64.
        v8f sacc[4] = {};
#pragma unroll
        for (int j = 0; j < 4; ++j) {
            const _Float16* krow = &Ks[(j * 16 + l15) * LDS2];
            sacc[j] = wmma_f16(qa[0], frag16x32(krow, h),      sacc[j]);
            sacc[j] = wmma_f16(qa[1], frag16x32(krow + 32, h), sacc[j]);
        }

        // Online softmax per row (row = t0 + h*8 + r; held by one 16-lane half).
#pragma unroll
        for (int r = 0; r < 8; ++r) {
            float mx = fmaxf(fmaxf(sacc[0][r], sacc[1][r]),
                             fmaxf(sacc[2][r], sacc[3][r]));
#pragma unroll
            for (int m = 8; m >= 1; m >>= 1) mx = fmaxf(mx, __shfl_xor(mx, m, 32));
            float newm  = fmaxf(rmax[r], mx);
            float scale = __expf(rmax[r] - newm);
            float p0 = __expf(sacc[0][r] - newm);
            float p1 = __expf(sacc[1][r] - newm);
            float p2 = __expf(sacc[2][r] - newm);
            float p3 = __expf(sacc[3][r] - newm);
            float ps = (p0 + p1) + (p2 + p3);
#pragma unroll
            for (int m = 8; m >= 1; m >>= 1) ps += __shfl_xor(ps, m, 32);
            rsum[r] = rsum[r] * scale + ps;
            rmax[r] = newm;
#pragma unroll
            for (int tj = 0; tj < 4; ++tj) cacc[tj][r] *= scale;
            _Float16* prow = &Pw[(h * 8 + r) * LDS2];
            prow[l15]      = (_Float16)p0;   // C-layout -> LDS
            prow[16 + l15] = (_Float16)p1;
            prow[32 + l15] = (_Float16)p2;
            prow[48 + l15] = (_Float16)p3;
        }

        // ctx += P(16x64) @ V(64x64); P re-read as two A-fragments (same-wave
        // LDS ops are in-order, no barrier needed for the per-wave P tile).
        v16h pa0 = frag16x32(&Pw[l15 * LDS2], h);
        v16h pa1 = frag16x32(&Pw[l15 * LDS2 + 32], h);
#pragma unroll
        for (int tj = 0; tj < 4; ++tj) {
            const _Float16* vrow = &Vt[(tj * 16 + l15) * LDS2];
            cacc[tj] = wmma_f16(pa0, frag16x32(vrow, h),      cacc[tj]);
            cacc[tj] = wmma_f16(pa1, frag16x32(vrow + 32, h), cacc[tj]);
        }
    }

    // Normalize and store ctx as f16 [b*2048 + t][n*64 + d].
#pragma unroll
    for (int tj = 0; tj < 4; ++tj)
#pragma unroll
        for (int r = 0; r < 8; ++r) {
            int row = bi * 2048 + t0 + h * 8 + r;
            int col = nh * 64 + tj * 16 + l15;
            ctxb[(size_t)row * 1024 + col] = (_Float16)(cacc[tj][r] / rsum[r]);
        }
}

// ---------------------------------------------------------------------------
// Kernel 3: output projection. ctx[8192,1024] (f16) @ Wout[1024,1024] (f32),
// scaled by 1/sqrt(64) = 0.125, fp32 output.  Same tiling as kernel 1.
// ---------------------------------------------------------------------------
__global__ __launch_bounds__(256) void out_gemm_kernel(
    const _Float16* __restrict__ CTX, const float* __restrict__ Wo,
    float* __restrict__ OUT)
{
    __shared__ _Float16 As[64 * LDA];
    __shared__ _Float16 Bts[256 * LDB];

    const int tid  = threadIdx.x;
    const int wave = tid >> 5, lane = tid & 31;
    const int l15  = lane & 15, h = lane >> 4;
    const int wm   = wave >> 2, wn = wave & 3;
    const int m0   = blockIdx.y * 64;
    const int n0   = blockIdx.x * 256;

    v8f acc[2][4] = {};

    for (int k0 = 0; k0 < 1024; k0 += 32) {
        if (k0 + 32 < 1024) {
            if (tid < 64)
                __builtin_prefetch(&CTX[(size_t)(m0 + tid) * 1024 + k0 + 32], 0, 0);
            __builtin_prefetch(&Wo[(size_t)(k0 + 32 + (tid >> 3)) * 1024 + n0 + (tid & 7) * 32], 0, 0);
        }
        __syncthreads();
#pragma unroll
        for (int e = 0; e < 8; ++e) {
            int lin = tid + e * 256;
            int r = lin >> 5, c = lin & 31;
            As[r * LDA + c] = CTX[(size_t)(m0 + r) * 1024 + k0 + c];
        }
#pragma unroll
        for (int e = 0; e < 32; ++e) {
            int lin = tid + e * 256;
            int kk = lin >> 8, n = lin & 255;
            Bts[n * LDB + kk] = (_Float16)Wo[(size_t)(k0 + kk) * 1024 + n0 + n];
        }
        __syncthreads();

        v16h a0 = frag16x32(&As[(wm * 32 + l15) * LDA], h);
        v16h a1 = frag16x32(&As[(wm * 32 + 16 + l15) * LDA], h);
#pragma unroll
        for (int tj = 0; tj < 4; ++tj) {
            v16h bf = frag16x32(&Bts[(wn * 64 + tj * 16 + l15) * LDB], h);
            acc[0][tj] = wmma_f16(a0, bf, acc[0][tj]);
            acc[1][tj] = wmma_f16(a1, bf, acc[1][tj]);
        }
    }

#pragma unroll
    for (int ti = 0; ti < 2; ++ti)
#pragma unroll
        for (int tj = 0; tj < 4; ++tj)
#pragma unroll
            for (int r = 0; r < 8; ++r) {
                int row = m0 + wm * 32 + ti * 16 + h * 8 + r;
                int col = n0 + wn * 64 + tj * 16 + l15;
                OUT[(size_t)row * 1024 + col] = acc[ti][tj][r] * 0.125f;
            }
}

// ---------------------------------------------------------------------------
extern "C" void kernel_launch(void* const* d_in, const int* in_sizes, int n_in,
                              void* d_out, int out_size, void* d_ws, size_t ws_size,
                              hipStream_t stream)
{
    const float* x   = (const float*)d_in[0];   // [4,2048,1024]
    const float* qkv = (const float*)d_in[1];   // [1024,3072]
    const float* wo  = (const float*)d_in[2];   // [1024,1024]
    float* out = (float*)d_out;                 // [4,2048,1024]

    const size_t HEAD_ELEMS = (size_t)4 * 16 * 2048 * 64;   // 8,388,608
    _Float16* qb   = (_Float16*)d_ws;
    _Float16* kb   = qb + HEAD_ELEMS;
    _Float16* vb   = kb + HEAD_ELEMS;
    _Float16* ctxb = vb + HEAD_ELEMS;                        // 8192 x 1024

    // 1) QKV projection: M=8192 (128 tiles of 64), N=3072 (12 tiles of 256)
    qkv_gemm_kernel<<<dim3(12, 128), 256, 0, stream>>>(x, qkv, qb, kb, vb);

    // 2) Flash attention: 64 heads x 16 query tiles
    flash_attn_kernel<<<dim3(1024), 256, 0, stream>>>(qb, kb, vb, ctxb);

    // 3) Output projection (+ 1/sqrt(d)): M=8192, N=1024 (4 tiles of 256)
    out_gemm_kernel<<<dim3(4, 128), 256, 0, stream>>>(ctxb, wo, out);
}
